// GL_TAGConv_9l_512h_w_k3_44753559224345
// MI455X (gfx1250) — matrix operand order repaired and checked by reading, not compiled, not run on hardware
//
#include <hip/hip_runtime.h>
#include <hip/hip_bf16.h>
#include <cstdint>

typedef __attribute__((ext_vector_type(16))) __bf16 bf16x16;
typedef __attribute__((ext_vector_type(8)))  float  f32x8;

// ---------------- elementwise / graph-norm kernels ----------------

__global__ void k_zero_i(int* __restrict__ p, int n) {
  int i = blockIdx.x * blockDim.x + threadIdx.x;
  if (i < n) p[i] = 0;
}

__global__ void k_count(const int* __restrict__ col, int* __restrict__ cnt, int E) {
  int i = blockIdx.x * blockDim.x + threadIdx.x;
  if (i < E) atomicAdd(&cnt[col[i]], 1);
}

__global__ void k_degw(const float* __restrict__ w, const int* __restrict__ col,
                       float* __restrict__ deg, int E) {
  int i = blockIdx.x * blockDim.x + threadIdx.x;
  if (i < E) atomicAdd(&deg[col[i]], w[i]);
}

__global__ void k_zero_f(float* __restrict__ p, int n) {
  int i = blockIdx.x * blockDim.x + threadIdx.x;
  if (i < n) p[i] = 0.0f;
}

__global__ void k_dinv(float* __restrict__ deg, int n) {
  int i = blockIdx.x * blockDim.x + threadIdx.x;
  if (i < n) { float d = deg[i]; deg[i] = d > 0.f ? rsqrtf(d) : 0.f; }
}

__global__ void k_norm(const float* __restrict__ w, const int* __restrict__ row,
                       const int* __restrict__ col, const float* __restrict__ dinv,
                       float* __restrict__ nrm, int E) {
  int i = blockIdx.x * blockDim.x + threadIdx.x;
  if (i < E) nrm[i] = dinv[row[i]] * w[i] * dinv[col[i]];
}

__global__ void k_elu(const float* __restrict__ in, float* __restrict__ out, int n) {
  int i = blockIdx.x * blockDim.x + threadIdx.x;
  if (i < n) {
    float v = in[i];
    out[i] = v > 0.f ? v : (__expf(v) - 1.f);
  }
}

// ---------------- CSR build (once per launch) ----------------
// One-block hierarchical exclusive scan of cnt[0..n) -> off[0..n], off[n]=total.
#define SCAN_T 1024
__global__ __launch_bounds__(SCAN_T)
void k_scan(const int* __restrict__ cnt, int* __restrict__ off, int n) {
  __shared__ int part[SCAN_T];
  int t = threadIdx.x;
  int chunk = (n + SCAN_T - 1) / SCAN_T;
  int begin = t * chunk;
  int end   = begin + chunk; if (end > n) end = n;
  int s = 0;
  for (int i = begin; i < end; ++i) s += cnt[i];
  part[t] = s;
  __syncthreads();
  for (int d = 1; d < SCAN_T; d <<= 1) {   // Hillis-Steele inclusive scan
    int v = (t >= d) ? part[t - d] : 0;
    __syncthreads();
    part[t] += v;
    __syncthreads();
  }
  int run = (t == 0) ? 0 : part[t - 1];
  for (int i = begin; i < end; ++i) { off[i] = run; run += cnt[i]; }
  if (t == SCAN_T - 1) off[n] = run;
}

// Slot fill: also pre-gathers per-slot source row and edge norm.
__global__ void k_fill(const int* __restrict__ row, const int* __restrict__ col,
                       const float* __restrict__ nrm, const int* __restrict__ off,
                       int* __restrict__ cursor, int* __restrict__ srow,
                       float* __restrict__ snrm, int E) {
  int e = blockIdx.x * blockDim.x + threadIdx.x;
  if (e < E) {
    int c = col[e];
    int p = atomicAdd(&cursor[c], 1);
    int s = off[c] + p;
    srow[s] = row[e];
    snrm[s] = nrm[e];
  }
}

// ---------------- SpMM gather: out[c,:] = sum_in-edges nrm * x[row,:] ----------------
// One block per destination node. Plain coalesced float4 loads, no atomics.
__global__ __launch_bounds__(128)
void k_spmm_gather(const float* __restrict__ x, const int* __restrict__ off,
                   const int* __restrict__ srow, const float* __restrict__ snrm,
                   float* __restrict__ out, int F) {
  int c = blockIdx.x;
  int s0 = off[c], s1 = off[c + 1];
  int nq = F >> 2;
  for (int q = threadIdx.x; q < nq; q += 128) {
    float4 acc = make_float4(0.f, 0.f, 0.f, 0.f);
    for (int s = s0; s < s1; ++s) {
      int r = srow[s];
      float w = snrm[s];
      float4 v = ((const float4*)(x + (size_t)r * F))[q];
      acc.x += w * v.x; acc.y += w * v.y; acc.z += w * v.z; acc.w += w * v.w;
    }
    ((float4*)(out + (size_t)c * F))[q] = acc;
  }
}

// ---------------- fragment packing (split fp32 -> bf16 hi/lo, WMMA layout) ----------------
// A: fp32 [M x K] row-major -> (16M x 32K) blocks, id = mT*(K/32)+kT, 512 bf16/block,
// lane l: row = mT*16+(l&15); koff=(l>>4)*16; elem i: k=koff+i
__global__ void k_pack_a(const float* __restrict__ x, __bf16* __restrict__ hi,
                         __bf16* __restrict__ lo, int M, int K) {
  int t = blockIdx.x * blockDim.x + threadIdx.x;
  int nK = K >> 5;
  int total = (M >> 4) * nK * 32;
  if (t >= total) return;
  int lane = t & 31;
  int blk  = t >> 5;
  int kT = blk % nK;
  int mT = blk / nK;
  int row  = mT * 16 + (lane & 15);
  int koff = (lane >> 4) * 16;
  const float* src = x + (size_t)row * K + kT * 32;
  __bf16* ph = hi + (size_t)blk * 512 + lane * 16;
  __bf16* pl = lo + (size_t)blk * 512 + lane * 16;
#pragma unroll
  for (int i = 0; i < 8; ++i) {
    float v0 = src[koff + i];
    float v1 = src[koff + 8 + i];
    __bf16 h0 = (__bf16)v0, h1 = (__bf16)v1;
    ph[i]     = h0;  pl[i]     = (__bf16)(v0 - (float)h0);
    ph[8 + i] = h1;  pl[8 + i] = (__bf16)(v1 - (float)h1);
  }
}

// B: fp32 [K x N] row-major -> (32K x 16N) blocks, id = nT*(K/32)+kT,
// lane l: col = nT*16+(l&15); elem j: k = (l>>4)*16 + j
__global__ void k_pack_b(const float* __restrict__ w, __bf16* __restrict__ hi,
                         __bf16* __restrict__ lo, int K, int N) {
  int t = blockIdx.x * blockDim.x + threadIdx.x;
  int nK = K >> 5;
  int total = (N >> 4) * nK * 32;
  if (t >= total) return;
  int lane = t & 31;
  int blk  = t >> 5;
  int kT = blk % nK;
  int nT = blk / nK;
  int c     = nT * 16 + (lane & 15);
  int kbase = kT * 32 + (lane >> 4) * 16;
  __bf16* ph = hi + (size_t)blk * 512 + lane * 16;
  __bf16* pl = lo + (size_t)blk * 512 + lane * 16;
#pragma unroll
  for (int j = 0; j < 16; ++j) {
    float v = w[(size_t)(kbase + j) * N + c];
    __bf16 h = (__bf16)v;
    ph[j] = h;
    pl[j] = (__bf16)(v - (float)h);
  }
}

// ---------------- WMMA GEMM, 2x2 register-blocked ----------------
// Each wave owns a 32(M) x 32(N) output patch: 4 accumulators, 12 WMMAs per 16 b128 loads.
__global__ __launch_bounds__(256)
void k_gemm_packed(const __bf16* __restrict__ Ahi, const __bf16* __restrict__ Alo,
                   const __bf16* __restrict__ Bhi, const __bf16* __restrict__ Blo,
                   const float* __restrict__ bias, float* __restrict__ C,
                   int N, int K, int accumulate, int mTiles) {
  const int lane = threadIdx.x & 31;
  const int wave = threadIdx.x >> 5;
  const int nTiles = N >> 4;
  const int n0 = (blockIdx.y * 8 + wave) * 2;
  if (n0 >= nTiles) return;                 // wave-uniform
  const int m0 = blockIdx.x * 2;
  const int m1 = m0 + 1;
  const bool m1ok = (m1 < mTiles);
  const int m1c = m1ok ? m1 : (mTiles - 1); // clamp: loads safe, stores guarded
  const int n1 = n0 + 1;
  const bool n1ok = (n1 < nTiles);
  const int n1c = n1ok ? n1 : (nTiles - 1);
  const int nK = K >> 5;
  const int half = lane >> 4;
  const int l15  = lane & 15;

  const int r0 = m0 * 16 + half * 8;
  const int r1 = m1c * 16 + half * 8;
  const int c0 = n0 * 16 + l15;
  const int c1 = n1c * 16 + l15;

  f32x8 acc00 = {}, acc01 = {}, acc10 = {}, acc11 = {};
  if (accumulate) {
#pragma unroll
    for (int i = 0; i < 8; ++i) {
      acc00[i] = C[(size_t)(r0 + i) * N + c0];
      acc01[i] = C[(size_t)(r0 + i) * N + c1];
      acc10[i] = C[(size_t)(r1 + i) * N + c0];
      acc11[i] = C[(size_t)(r1 + i) * N + c1];
    }
  }
  {
    float bv0 = bias[c0], bv1 = bias[c1];
#pragma unroll
    for (int i = 0; i < 8; ++i) {
      acc00[i] += bv0; acc01[i] += bv1; acc10[i] += bv0; acc11[i] += bv1;
    }
  }

  // uint4 units: one 512-bf16 fragment block = 64 uint4; one lane chunk = 2 uint4
  const uint4* pA0h = (const uint4*)Ahi + (size_t)(m0  * nK) * 64 + lane * 2;
  const uint4* pA0l = (const uint4*)Alo + (size_t)(m0  * nK) * 64 + lane * 2;
  const uint4* pA1h = (const uint4*)Ahi + (size_t)(m1c * nK) * 64 + lane * 2;
  const uint4* pA1l = (const uint4*)Alo + (size_t)(m1c * nK) * 64 + lane * 2;
  const uint4* pB0h = (const uint4*)Bhi + (size_t)(n0  * nK) * 64 + lane * 2;
  const uint4* pB0l = (const uint4*)Blo + (size_t)(n0  * nK) * 64 + lane * 2;
  const uint4* pB1h = (const uint4*)Bhi + (size_t)(n1c * nK) * 64 + lane * 2;
  const uint4* pB1l = (const uint4*)Blo + (size_t)(n1c * nK) * 64 + lane * 2;

  for (int kT = 0; kT < nK; ++kT) {
    union { uint4 u[2]; bf16x16 v; } a0h, a0l, a1h, a1l, b0h, b0l, b1h, b1l;
    a0h.u[0] = pA0h[0]; a0h.u[1] = pA0h[1];
    a0l.u[0] = pA0l[0]; a0l.u[1] = pA0l[1];
    a1h.u[0] = pA1h[0]; a1h.u[1] = pA1h[1];
    a1l.u[0] = pA1l[0]; a1l.u[1] = pA1l[1];
    b0h.u[0] = pB0h[0]; b0h.u[1] = pB0h[1];
    b0l.u[0] = pB0l[0]; b0l.u[1] = pB0l[1];
    b1h.u[0] = pB1h[0]; b1h.u[1] = pB1h[1];
    b1l.u[0] = pB1l[0]; b1l.u[1] = pB1l[1];
    pA0h += 64; pA0l += 64; pA1h += 64; pA1l += 64;
    pB0h += 64; pB0l += 64; pB1h += 64; pB1l += 64;

    // bf16x3 per tile-pair: hi*hi + hi*lo + lo*hi
    acc00 = __builtin_amdgcn_wmma_f32_16x16x32_bf16(false, a0h.v, false, b0h.v, (short)0, acc00, false, false);
    acc00 = __builtin_amdgcn_wmma_f32_16x16x32_bf16(false, a0h.v, false, b0l.v, (short)0, acc00, false, false);
    acc00 = __builtin_amdgcn_wmma_f32_16x16x32_bf16(false, a0l.v, false, b0h.v, (short)0, acc00, false, false);

    acc01 = __builtin_amdgcn_wmma_f32_16x16x32_bf16(false, a0h.v, false, b1h.v, (short)0, acc01, false, false);
    acc01 = __builtin_amdgcn_wmma_f32_16x16x32_bf16(false, a0h.v, false, b1l.v, (short)0, acc01, false, false);
    acc01 = __builtin_amdgcn_wmma_f32_16x16x32_bf16(false, a0l.v, false, b1h.v, (short)0, acc01, false, false);

    acc10 = __builtin_amdgcn_wmma_f32_16x16x32_bf16(false, a1h.v, false, b0h.v, (short)0, acc10, false, false);
    acc10 = __builtin_amdgcn_wmma_f32_16x16x32_bf16(false, a1h.v, false, b0l.v, (short)0, acc10, false, false);
    acc10 = __builtin_amdgcn_wmma_f32_16x16x32_bf16(false, a1l.v, false, b0h.v, (short)0, acc10, false, false);

    acc11 = __builtin_amdgcn_wmma_f32_16x16x32_bf16(false, a1h.v, false, b1h.v, (short)0, acc11, false, false);
    acc11 = __builtin_amdgcn_wmma_f32_16x16x32_bf16(false, a1h.v, false, b1l.v, (short)0, acc11, false, false);
    acc11 = __builtin_amdgcn_wmma_f32_16x16x32_bf16(false, a1l.v, false, b1h.v, (short)0, acc11, false, false);
  }

#pragma unroll
  for (int i = 0; i < 8; ++i) C[(size_t)(r0 + i) * N + c0] = acc00[i];
  if (n1ok) {
#pragma unroll
    for (int i = 0; i < 8; ++i) C[(size_t)(r0 + i) * N + c1] = acc01[i];
  }
  if (m1ok) {
#pragma unroll
    for (int i = 0; i < 8; ++i) C[(size_t)(r1 + i) * N + c0] = acc10[i];
  }
  if (m1ok && n1ok) {
#pragma unroll
    for (int i = 0; i < 8; ++i) C[(size_t)(r1 + i) * N + c1] = acc11[i];
  }
}

// ---------------- host orchestration ----------------

static inline int ceil_div(int a, int b) { return (a + b - 1) / b; }

extern "C" void kernel_launch(void* const* d_in, const int* in_sizes, int n_in,
                              void* d_out, int out_size, void* d_ws, size_t ws_size,
                              hipStream_t stream) {
  (void)in_sizes; (void)n_in; (void)out_size; (void)ws_size;
  constexpr int NN = 10000, NE = 160000, NF = 128, NH = 512, NC = 16, NL = 9;

  const float* x  = (const float*)d_in[0];
  const float* ew = (const float*)d_in[1];
  const float* Wl[NL];
  const float* Bl[NL];
  for (int l = 0; l < NL; ++l) {
    Wl[l] = (const float*)d_in[2 + 2 * l];   // [4, din, dout]
    Bl[l] = (const float*)d_in[3 + 2 * l];   // [4, dout]
  }
  const int* ei   = (const int*)d_in[2 + 2 * NL]; // [2, NE]
  const int* erow = ei;
  const int* ecol = ei + NE;

  int din[NL], dout[NL];
  size_t woff[NL], wtot = 0;
  for (int l = 0; l < NL; ++l) {
    din[l]  = (l == 0) ? NF : NH;
    dout[l] = (l == NL - 1) ? NC : NH;
    woff[l] = wtot;
    wtot += (size_t)4 * din[l] * dout[l];
  }

  // carve workspace
  uintptr_t wp = (uintptr_t)d_ws;
  auto carve = [&](size_t bytes) -> void* {
    void* r = (void*)wp;
    wp += (bytes + 255) & ~(size_t)255;
    return r;
  };
  float*  deg  = (float*)carve((size_t)NN * 4);
  float*  nrm  = (float*)carve((size_t)NE * 4);
  int*    cnt  = (int*)carve((size_t)NN * 4);
  int*    off  = (int*)carve((size_t)(NN + 1) * 4);
  int*    cur  = (int*)carve((size_t)NN * 4);
  int*    srow = (int*)carve((size_t)NE * 4);
  float*  snrm = (float*)carve((size_t)NE * 4);
  float*  X0   = (float*)carve((size_t)NN * NH * 4);
  float*  X1   = (float*)carve((size_t)NN * NH * 4);
  float*  X2   = (float*)carve((size_t)NN * NH * 4);
  float*  OUT  = (float*)carve((size_t)NN * NH * 4);
  __bf16* aHi  = (__bf16*)carve((size_t)NN * NH * 2);
  __bf16* aLo  = (__bf16*)carve((size_t)NN * NH * 2);
  __bf16* wHi  = (__bf16*)carve(wtot * 2);
  __bf16* wLo  = (__bf16*)carve(wtot * 2);

  // gcn_norm: deg = segsum(w over col); dinv = rsqrt; norm = dinv[row]*w*dinv[col]
  k_zero_f<<<ceil_div(NN, 256), 256, 0, stream>>>(deg, NN);
  k_degw<<<ceil_div(NE, 256), 256, 0, stream>>>(ew, ecol, deg, NE);
  k_dinv<<<ceil_div(NN, 256), 256, 0, stream>>>(deg, NN);
  k_norm<<<ceil_div(NE, 256), 256, 0, stream>>>(ew, erow, ecol, deg, nrm, NE);

  // CSR build keyed by destination (col)
  k_zero_i<<<ceil_div(NN, 256), 256, 0, stream>>>(cnt, NN);
  k_count<<<ceil_div(NE, 256), 256, 0, stream>>>(ecol, cnt, NE);
  k_scan<<<1, SCAN_T, 0, stream>>>(cnt, off, NN);
  k_zero_i<<<ceil_div(NN, 256), 256, 0, stream>>>(cur, NN);
  k_fill<<<ceil_div(NE, 256), 256, 0, stream>>>(erow, ecol, nrm, off, cur, srow, snrm, NE);

  // pack all weight hops into WMMA B-fragment order (bf16 hi/lo)
  for (int l = 0; l < NL; ++l) {
    for (int k = 0; k < 4; ++k) {
      size_t woffk = woff[l] + (size_t)k * din[l] * dout[l];
      int nthreads = (dout[l] >> 4) * (din[l] >> 5) * 32;
      k_pack_b<<<ceil_div(nthreads, 256), 256, 0, stream>>>(
          Wl[l] + (size_t)k * din[l] * dout[l], wHi + woffk, wLo + woffk, din[l], dout[l]);
    }
  }

  const int mTiles = NN >> 4;  // 625
  const float* h = x;
  for (int l = 0; l < NL; ++l) {
    const int Kd = din[l], Nd = dout[l];
    float* out = (l == NL - 1) ? (float*)d_out : OUT;
    dim3 ggrid(ceil_div(mTiles, 2), ceil_div(ceil_div(Nd >> 4, 2), 8));
    const int packThreads = mTiles * (Kd >> 5) * 32;

    // hop 0: out = h @ W0 + b0
    k_pack_a<<<ceil_div(packThreads, 256), 256, 0, stream>>>(h, aHi, aLo, NN, Kd);
    k_gemm_packed<<<ggrid, 256, 0, stream>>>(aHi, aLo, wHi + woff[l], wLo + woff[l],
                                             Bl[l], out, Nd, Kd, 0, mTiles);

    // hops 1..3: p_k = A_norm @ p_{k-1}; out += p_k @ Wk + bk
    const float* src = h;
    float* pbuf[3] = {X1, X2, X0};  // X0 (current h) safely recycled for hop 3
    for (int k = 1; k <= 3; ++k) {
      float* pk = pbuf[k - 1];
      k_spmm_gather<<<NN, 128, 0, stream>>>(src, off, srow, snrm, pk, Kd);
      k_pack_a<<<ceil_div(packThreads, 256), 256, 0, stream>>>(pk, aHi, aLo, NN, Kd);
      size_t wo = woff[l] + (size_t)k * Kd * Nd;
      k_gemm_packed<<<ggrid, 256, 0, stream>>>(aHi, aLo, wHi + wo, wLo + wo,
                                               Bl[l] + (size_t)k * Nd, out, Nd, Kd, 1, mTiles);
      src = pk;
    }

    if (l < NL - 1) {
      int n = NN * Nd;
      k_elu<<<ceil_div(n, 256), 256, 0, stream>>>(out, X0, n);
      h = X0;
    }
  }
}
